// GATLayerV2_39694087750291
// MI455X (gfx1250) — matrix operand order repaired and checked by reading, not compiled, not run on hardware
//
#include <hip/hip_runtime.h>

#define NN 50000
#define EE 800000
#define FIN 128
#define HH 4
#define CC 32
#define HC 128
#define EDIM 16
#define NEG_SLOPE 0.2f

typedef __attribute__((ext_vector_type(16))) __bf16 v16bf;
typedef __attribute__((ext_vector_type(8)))  float  v8f;

// ---- monotone uint key for float atomic-max ----
__device__ __forceinline__ unsigned fkey(float f) {
  unsigned u = __float_as_uint(f);
  return (u & 0x80000000u) ? ~u : (u | 0x80000000u);
}
__device__ __forceinline__ float funkey(unsigned k) {
  return __uint_as_float((k & 0x80000000u) ? (k & 0x7FFFFFFFu) : ~k);
}
__device__ __forceinline__ void atomAddF(float* p, float v) {
  __hip_atomic_fetch_add(p, v, __ATOMIC_RELAXED, __HIP_MEMORY_SCOPE_AGENT);
}

// ---------------------------------------------------------------------------
// 0) init: zero the output accumulator + denom, set smax keys to key(-inf)
// ---------------------------------------------------------------------------
__global__ void gat_init(float* __restrict__ out, float* __restrict__ denom,
                         unsigned* __restrict__ smax) {
  int t = blockIdx.x * blockDim.x + threadIdx.x;
  if (t < NN * HC) out[t] = 0.0f;
  if (t < NN * HH) { denom[t] = 0.0f; smax[t] = 0x007FFFFFu; }  // key(-inf)
}

// ---------------------------------------------------------------------------
// 1) fused node GEMMs via WMMA bf16: xl = x@Wl^T+bl, xr = x@Wr^T+br,
//    skipb = x@Wskip^T + bias.  Block = 256 thr = 8 waves; block owns 16 nodes,
//    wave w owns output cols [16w,16w+16) of all three matrices.
// ---------------------------------------------------------------------------
__global__ void __launch_bounds__(256)
node_gemm(const float* __restrict__ x,
          const float* __restrict__ Wl, const float* __restrict__ bl,
          const float* __restrict__ Wr, const float* __restrict__ br,
          const float* __restrict__ Wsk, const float* __restrict__ bias,
          float* __restrict__ xl, float* __restrict__ xr,
          float* __restrict__ skipb) {
  const int wave = threadIdx.x >> 5;
  const int lane = threadIdx.x & 31;
  const int node0 = blockIdx.x * 16;
  const int lmod  = lane & 15;
  const int ocol  = wave * 16 + lmod;            // output column this lane owns
  const int m     = node0 + lmod;                // A-matrix row (node)
  const int kloA  = (lane < 16) ? 0 : 16;        // A frag K base (contiguous, matches B)
  const int kloB  = (lane < 16) ? 0 : 16;        // B frag K base (contiguous, matches A)

  v8f accL = {}, accR = {}, accS = {};

  for (int kk = 0; kk < FIN; kk += 32) {
    // A fragment: x[m][kk+kloA + {0..15}]
    v16bf a;
    const float* xrow = x + (size_t)m * FIN + kk + kloA;
#pragma unroll
    for (int i = 0; i < 8; ++i) {
      a[i]     = (__bf16)xrow[i];
      a[i + 8] = (__bf16)xrow[8 + i];
    }
    // B fragments: B[k][ocol] = W[ocol][k] -> 16 contiguous floats of row ocol
    const float* wl = Wl  + (size_t)ocol * FIN + kk + kloB;
    const float* wr = Wr  + (size_t)ocol * FIN + kk + kloB;
    const float* ws = Wsk + (size_t)ocol * FIN + kk + kloB;
    v16bf bL, bR, bS;
#pragma unroll
    for (int i = 0; i < 16; ++i) {
      bL[i] = (__bf16)wl[i];
      bR[i] = (__bf16)wr[i];
      bS[i] = (__bf16)ws[i];
    }
    accL = __builtin_amdgcn_wmma_f32_16x16x32_bf16(false, a, false, bL,
                                                   (short)0, accL, false, false);
    accR = __builtin_amdgcn_wmma_f32_16x16x32_bf16(false, a, false, bR,
                                                   (short)0, accR, false, false);
    accS = __builtin_amdgcn_wmma_f32_16x16x32_bf16(false, a, false, bS,
                                                   (short)0, accS, false, false);
  }

  const float bLv = bl[ocol], bRv = br[ocol], bBv = bias[ocol];
  const int row0 = node0 + ((lane < 16) ? 0 : 8);   // C/D layout: VGPR r -> M=r(+8)
#pragma unroll
  for (int r = 0; r < 8; ++r) {
    size_t o = (size_t)(row0 + r) * HC + ocol;
    xl[o]    = accL[r] + bLv;
    xr[o]    = accR[r] + bRv;
    skipb[o] = accS[r] + bBv;
  }
}

// ---------------------------------------------------------------------------
// 2) per-edge score: z = leakyrelu(hs+hd+ef), score = <z, att>, atomic max.
//    One wave per edge; lane l owns channels 4l..4l+3; head h = l/8.
// ---------------------------------------------------------------------------
__global__ void __launch_bounds__(256)
edge_score(const long long* __restrict__ ei, const float* __restrict__ eattr,
           const float* __restrict__ We, const float* __restrict__ att,
           const float* __restrict__ xl, const float* __restrict__ xr,
           float* __restrict__ score, unsigned* __restrict__ smax) {
  __shared__ float sWe[HC * 17];  // stride 17: no bank conflicts
  for (int i = threadIdx.x; i < HC * EDIM; i += blockDim.x) {
    int c = i / EDIM, k = i % EDIM;
    sWe[c * 17 + k] = We[i];
  }
  __syncthreads();

  const int wave = threadIdx.x >> 5;
  const int lane = threadIdx.x & 31;
  const int e = blockIdx.x * 8 + wave;
  if (e >= EE) return;

  const int src = (int)ei[e];
  const int dst = (int)ei[EE + e];
  const int c0  = lane * 4;

  const float4 hs = *(const float4*)(xl + (size_t)src * HC + c0);
  const float4 hd = *(const float4*)(xr + (size_t)dst * HC + c0);

  // ef[c] = sum_k edge_attr[e][k] * We[c][k]  (K = 16, in-register)
  float ef0 = 0.f, ef1 = 0.f, ef2 = 0.f, ef3 = 0.f;
  const float* ea = eattr + (size_t)e * EDIM;
#pragma unroll
  for (int k = 0; k < EDIM; ++k) {
    const float av = ea[k];  // same address across lanes -> broadcast
    ef0 = fmaf(av, sWe[(c0 + 0) * 17 + k], ef0);
    ef1 = fmaf(av, sWe[(c0 + 1) * 17 + k], ef1);
    ef2 = fmaf(av, sWe[(c0 + 2) * 17 + k], ef2);
    ef3 = fmaf(av, sWe[(c0 + 3) * 17 + k], ef3);
  }

  float z0 = hs.x + hd.x + ef0; z0 = (z0 > 0.f) ? z0 : NEG_SLOPE * z0;
  float z1 = hs.y + hd.y + ef1; z1 = (z1 > 0.f) ? z1 : NEG_SLOPE * z1;
  float z2 = hs.z + hd.z + ef2; z2 = (z2 > 0.f) ? z2 : NEG_SLOPE * z2;
  float z3 = hs.w + hd.w + ef3; z3 = (z3 > 0.f) ? z3 : NEG_SLOPE * z3;

  // att is [H,C] flat = 128 floats; channel c maps directly to att[c]
  float p = z0 * att[c0] + z1 * att[c0 + 1] + z2 * att[c0 + 2] + z3 * att[c0 + 3];
  p += __shfl_xor(p, 1, 32);
  p += __shfl_xor(p, 2, 32);
  p += __shfl_xor(p, 4, 32);   // reduced within each 8-lane head group

  if ((lane & 7) == 0) {
    const int h = lane >> 3;
    score[e * HH + h] = p;
    atomicMax(&smax[dst * HH + h], fkey(p));
  }
}

// ---------------------------------------------------------------------------
// 3) ex = exp(score - smax[dst]); denom[dst] += ex  (ex overwrites score)
// ---------------------------------------------------------------------------
__global__ void edge_softmax(const long long* __restrict__ ei,
                             const unsigned* __restrict__ smax,
                             float* __restrict__ score_ex,
                             float* __restrict__ denom) {
  int t = blockIdx.x * blockDim.x + threadIdx.x;
  if (t >= EE * HH) return;
  const int e = t >> 2, h = t & 3;
  const int dst = (int)ei[EE + e];
  const float m = funkey(smax[dst * HH + h]);
  const float ex = __expf(score_ex[t] - m);
  score_ex[t] = ex;
  atomAddF(&denom[dst * HH + h], ex);
}

// ---------------------------------------------------------------------------
// 4) scatter: out[dst] += alpha * hs.  One wave per edge, 4 ch per lane.
// ---------------------------------------------------------------------------
__global__ void __launch_bounds__(256)
edge_scatter(const long long* __restrict__ ei, const float* __restrict__ xl,
             const float* __restrict__ ex, const float* __restrict__ denom,
             float* __restrict__ out) {
  const int wave = threadIdx.x >> 5;
  const int lane = threadIdx.x & 31;
  const int e = blockIdx.x * 8 + wave;
  if (e >= EE) return;

  const int src = (int)ei[e];
  const int dst = (int)ei[EE + e];
  const int c0  = lane * 4;
  const int h   = lane >> 3;

  const float alpha = ex[e * HH + h] / (denom[dst * HH + h] + 1e-16f);
  const float4 hs = *(const float4*)(xl + (size_t)src * HC + c0);
  float* o = out + (size_t)dst * HC + c0;
  atomAddF(o + 0, alpha * hs.x);
  atomAddF(o + 1, alpha * hs.y);
  atomAddF(o + 2, alpha * hs.z);
  atomAddF(o + 3, alpha * hs.w);
}

// ---------------------------------------------------------------------------
// 5) finalize: out = elu(out + skip + bias)   (bias folded into skipb)
// ---------------------------------------------------------------------------
__global__ void gat_finalize(float* __restrict__ out,
                             const float* __restrict__ skipb) {
  int t = blockIdx.x * blockDim.x + threadIdx.x;
  if (t >= NN * HC) return;
  const float v = out[t] + skipb[t];
  out[t] = (v > 0.f) ? v : (__expf(v) - 1.0f);
}

// ---------------------------------------------------------------------------
extern "C" void kernel_launch(void* const* d_in, const int* in_sizes, int n_in,
                              void* d_out, int out_size, void* d_ws,
                              size_t ws_size, hipStream_t stream) {
  (void)in_sizes; (void)n_in; (void)out_size; (void)ws_size;
  const float*     x     = (const float*)d_in[0];
  const long long* ei    = (const long long*)d_in[1];   // int64 [2,E]
  const float*     eattr = (const float*)d_in[2];
  const float*     Wl    = (const float*)d_in[3];
  const float*     bl    = (const float*)d_in[4];
  const float*     Wr    = (const float*)d_in[5];
  const float*     br    = (const float*)d_in[6];
  const float*     We    = (const float*)d_in[7];
  const float*     att   = (const float*)d_in[8];
  const float*     bias  = (const float*)d_in[9];
  const float*     Wsk   = (const float*)d_in[10];
  float* out = (float*)d_out;

  // workspace carve-out
  char* w = (char*)d_ws;
  float* xl    = (float*)w; w += (size_t)NN * HC * sizeof(float);
  float* xr    = (float*)w; w += (size_t)NN * HC * sizeof(float);
  float* skipb = (float*)w; w += (size_t)NN * HC * sizeof(float);
  float* score = (float*)w; w += (size_t)EE * HH * sizeof(float);  // becomes ex
  float* denom = (float*)w; w += (size_t)NN * HH * sizeof(float);
  unsigned* smax = (unsigned*)w;

  gat_init<<<(NN * HC + 255) / 256, 256, 0, stream>>>(out, denom, smax);
  node_gemm<<<NN / 16, 256, 0, stream>>>(x, Wl, bl, Wr, br, Wsk, bias,
                                         xl, xr, skipb);
  edge_score<<<EE / 8, 256, 0, stream>>>(ei, eattr, We, att, xl, xr,
                                         score, smax);
  edge_softmax<<<(EE * HH + 255) / 256, 256, 0, stream>>>(ei, smax, score,
                                                          denom);
  edge_scatter<<<EE / 8, 256, 0, stream>>>(ei, xl, score, denom, out);
  gat_finalize<<<(NN * HC + 255) / 256, 256, 0, stream>>>(out, skipb);
}
